// Fusion_Layer_58815282151552
// MI455X (gfx1250) — matrix operand order repaired and checked by reading, not compiled, not run on hardware
//
#include <hip/hip_runtime.h>
#include <hip/hip_bf16.h>
#include <math.h>

// CDNA5 / gfx1250, wave32, WMMA bf16 path.

typedef __attribute__((ext_vector_type(16))) __bf16 v16bf;
typedef __attribute__((ext_vector_type(8)))  float  v8f;

#define C_IN     256      // mid channels (K for main GEMM)
#define C_GLOB   256      // global channels
#define C_OUT    256      // N
#define HW       784      // 28*28 spatial positions per image
#define KB_COUNT 8        // 256 / 32 K-steps

// Fast tanh: gfx1250 has a hardware transcendental v_tanh_f32.
// Fall back to the exp2 identity (saturates correctly through inf/0).
__device__ __forceinline__ float fast_tanh(float x) {
#if __has_builtin(__builtin_amdgcn_tanhf)
    return __builtin_amdgcn_tanhf(x);
#else
    // tanh(x) = 1 - 2/(exp2(2*log2(e)*x) + 1)
    const float e = __builtin_amdgcn_exp2f(x * 2.8853900817779268f);
    return 1.0f - 2.0f * __builtin_amdgcn_rcpf(e + 1.0f);
#endif
}

// ---------------------------------------------------------------------------
// Prep 1: pack W_mid (rows k=0..255 of conv_w, layout W[k][n]) into this kernel's
// per-lane WMMA B-fragment layout for V_WMMA_F32_16X16X32_BF16 (contiguous k, same order as A):
//   lane 0..15  (col n = lane)  hold K = kb*32 + 0..15   (half = 0)
//   lane 16..31 (col n = l-16)  hold K = kb*32 + 16..31  (half = 1)
// packed so each lane's 16 bf16 are contiguous (one 32B load):
//   wpk[((kb*256 + col)*2 + half)*16 + i] = W[kb*32 + half*16 + i][col]
// ---------------------------------------------------------------------------
__global__ void __launch_bounds__(256) pack_wmid(const float* __restrict__ w,
                                                 __bf16* __restrict__ wpk) {
    int o   = blockIdx.x * blockDim.x + threadIdx.x;  // 0 .. 65535
    int i   = o & 15;
    int t   = o >> 4;
    int h   = t & 1;
    int col = (t >> 1) & 255;
    int kb  = t >> 9;
    int k   = kb * 32 + h * 16 + i;
    wpk[o] = (__bf16)w[k * C_OUT + col];
}

// ---------------------------------------------------------------------------
// Prep 2: fold the global-feature half of the GEMM into a per-image bias:
//   gbias[b][n] = conv_b[n] + sum_k global[b][k] * W[256+k][n]
// 128 blocks x 256 threads; coalesced across n. 16.8 MFLOP -> negligible.
// ---------------------------------------------------------------------------
__global__ void __launch_bounds__(256) make_gbias(const float* __restrict__ g,
                                                  const float* __restrict__ w,
                                                  const float* __restrict__ bias,
                                                  float* __restrict__ gb) {
    const int b = blockIdx.x;
    const int n = threadIdx.x;
    const float* gr = g + b * C_GLOB;
    float s = bias[n];
#pragma unroll 4
    for (int k = 0; k < C_GLOB; ++k)
        s += gr[k] * w[(C_IN + k) * C_OUT + n];
    gb[b * C_OUT + n] = s;
}

// ---------------------------------------------------------------------------
// Main GEMM: out[m][n] = tanh( mid[m][:] @ Wmid[:, n] + gbias[m/784][n] )
// M = 100352 (divisible by 64), K = 256, N = 256.
// Block = 256 threads = 8 waves; block tile = 64(M) x 256(N).
// Wave grid = 2(M-groups of 32 rows) x 4(N-groups of 64 cols):
//   per wave: 2 M-subtiles x 4 N-subtiles = 8 accumulators (v8f each).
// This balances redundancy: A fetched x4, B fetched x2 per block
// (A is f32, B is bf16 -> 64 KB cache traffic per 32-K chunk vs 80 KB
// for the 1x8 arrangement).
// ---------------------------------------------------------------------------
__global__ void __launch_bounds__(256) fusion_gemm(const float*  __restrict__ mid,
                                                   const __bf16* __restrict__ wpk,
                                                   const float*  __restrict__ gb,
                                                   float*        __restrict__ out) {
    const int lane = threadIdx.x & 31;
    const int wave = threadIdx.x >> 5;
    const int kh   = lane >> 4;      // K-half selector (lanes 16-31 take upper K)
    const int l15  = lane & 15;
    const int m0   = blockIdx.x * 64;
    const int wm   = wave & 1;       // M-group: rows [wm*32, wm*32+32)
    const int wn   = wave >> 1;      // N-group: cols [wn*64, wn*64+64)
    const int mb   = m0 + wm * 32;
    const int n0   = wn * 64;

    v8f acc[2][4] = {};

    // Per-lane A-row pointers for the 2 M-subtiles (A layout: M = l15).
    const float* arow[2];
#pragma unroll
    for (int mi = 0; mi < 2; ++mi)
        arow[mi] = mid + (size_t)(mb + mi * 16 + l15) * C_IN;

    // Per-lane B-fragment base pointers (advance by 8192 bf16 per K-step).
    const __bf16* bb[4];
#pragma unroll
    for (int nj = 0; nj < 4; ++nj)
        bb[nj] = wpk + ((size_t)(n0 + nj * 16 + l15) * 2 + kh) * 16;

#pragma unroll
    for (int kb = 0; kb < KB_COUNT; ++kb) {
        const int kbase = kb * 32 + kh * 16;  // lanes 0-15: K 0..15 ; 16-31: K 16..31

        v16bf a[2];
#pragma unroll
        for (int mi = 0; mi < 2; ++mi) {
            const float4 c0 = *(const float4*)(arow[mi] + kbase);       // K +0..3
            const float4 c1 = *(const float4*)(arow[mi] + kbase + 4);   // K +4..7
            const float4 c2 = *(const float4*)(arow[mi] + kbase + 8);   // K +8..11
            const float4 c3 = *(const float4*)(arow[mi] + kbase + 12);  // K +12..15
            v16bf av;
            av[0]  = (__bf16)c0.x; av[1]  = (__bf16)c0.y;
            av[2]  = (__bf16)c0.z; av[3]  = (__bf16)c0.w;
            av[4]  = (__bf16)c1.x; av[5]  = (__bf16)c1.y;
            av[6]  = (__bf16)c1.z; av[7]  = (__bf16)c1.w;
            av[8]  = (__bf16)c2.x; av[9]  = (__bf16)c2.y;
            av[10] = (__bf16)c2.z; av[11] = (__bf16)c2.w;
            av[12] = (__bf16)c3.x; av[13] = (__bf16)c3.y;
            av[14] = (__bf16)c3.z; av[15] = (__bf16)c3.w;
            a[mi] = av;
        }

        v16bf b[4];
#pragma unroll
        for (int nj = 0; nj < 4; ++nj)
            b[nj] = *(const v16bf*)(bb[nj] + (size_t)kb * 8192);

#pragma unroll
        for (int mi = 0; mi < 2; ++mi) {
#pragma unroll
            for (int nj = 0; nj < 4; ++nj) {
                acc[mi][nj] = __builtin_amdgcn_wmma_f32_16x16x32_bf16(
                    false, a[mi], false, b[nj], (short)0, acc[mi][nj], false, false);
            }
        }
    }

    // Epilogue. C/D layout: lanes 0-15 -> M = v, N = lane;
    //                       lanes 16-31 -> M = 8+v, N = lane-16.
#pragma unroll
    for (int mi = 0; mi < 2; ++mi) {
#pragma unroll
        for (int nj = 0; nj < 4; ++nj) {
            const int coln  = n0 + nj * 16 + l15;
            const int rbase = mb + mi * 16 + kh * 8;
#pragma unroll
            for (int v = 0; v < 8; ++v) {
                const int r    = rbase + v;
                const int bimg = r / HW;                       // magic-mul divide
                const float x  = acc[mi][nj][v] + gb[bimg * C_OUT + coln];
                out[(size_t)r * C_OUT + coln] = fast_tanh(x);
            }
        }
    }
}

// ---------------------------------------------------------------------------
// Inputs (setup_inputs order):
//   d_in[0] mid_features    [128,28,28,256] f32
//   d_in[1] global_features [128,256]       f32
//   d_in[2] conv_w          [1,1,512,256]   f32  (W[k][n])
//   d_in[3] conv_b          [256]           f32
//   d_in[4] stride          (==1, ignored; 1x1 conv SAME stride 1)
// Output: [128,28,28,256] f32
// Workspace: 128 KB packed bf16 weights + 128 KB f32 gbias = 256 KB
// ---------------------------------------------------------------------------
extern "C" void kernel_launch(void* const* d_in, const int* in_sizes, int n_in,
                              void* d_out, int out_size, void* d_ws, size_t ws_size,
                              hipStream_t stream) {
    const float* mid  = (const float*)d_in[0];
    const float* gf   = (const float*)d_in[1];
    const float* w    = (const float*)d_in[2];
    const float* bias = (const float*)d_in[3];
    float* out = (float*)d_out;

    __bf16* wpk = (__bf16*)d_ws;                                  // 65536 bf16
    float*  gb  = (float*)((char*)d_ws + 65536 * sizeof(__bf16)); // 32768 f32

    pack_wmid <<<256,  256, 0, stream>>>(w, wpk);
    make_gbias<<<128,  256, 0, stream>>>(gf, w, bias, gb);

    // M = 100352 rows / 64 per block = 1568 blocks (exact, no tail).
    fusion_gemm<<<1568, 256, 0, stream>>>(mid, wpk, gb, out);
}
